// MnistModelDP_71511205478930
// MI455X (gfx1250) — matrix operand order repaired and checked by reading, compiled, not run on hardware
//
#include <hip/hip_runtime.h>
#include <hip/hip_fp16.h>
#include <cstdint>
#include <cstddef>

// ---------------------------------------------------------------------------
// 2-layer LSTM (T=64, B=2048, IN=128, H=512) + linear head, for gfx1250.
// GEMMs on v_wmma_f32_16x16x32_f16 (f16 in, f32 acc). Each workgroup owns a
// 16-row batch tile; h (f16) / c (f32) live in LDS across the time loop.
// Next step's x tile is prefetched with CDNA5 async global->LDS copies
// (global_load_async_to_lds_b128 / s_wait_asynccnt) into a double buffer,
// hiding staging behind the WMMA phase.
// ---------------------------------------------------------------------------

typedef __attribute__((ext_vector_type(16))) _Float16 v16h;
typedef __attribute__((ext_vector_type(4)))  _Float16 v4h;
typedef __attribute__((ext_vector_type(8)))  float    v8f;

#define T_STEPS 64
#define BATCH   2048
#define HID     512
#define GATES   2048      // 4*HID
#define BT      16        // batch rows per workgroup (one WMMA M tile)
#define NW      8         // waves per workgroup
#define WTH     (NW * 32) // 256 threads

__device__ __forceinline__ float sigf(float x) {
    return 1.0f / (1.0f + __expf(-x));
}

// A-fragment (16x32 f16, M x K) from LDS, row-major rows of `ld` halves.
// Layout: lanes 0-15 = M; hi half-wave holds K = 16*hi + s (contiguous,
// matching B); each 32-bit VGPR holds an even/odd K pair.
__device__ __forceinline__ v16h load_a_frag(const _Float16* a, int ld, int kbase, int lane) {
    const int m  = lane & 15;
    const int hi = lane >> 4;
    union { v16h h; uint32_t u[8]; } f;
    const uint32_t* row = (const uint32_t*)(a + m * ld + kbase);
#pragma unroll
    for (int v = 0; v < 8; ++v) {
        const int k = 16 * hi + 2 * v;
        f.u[v] = row[k >> 1];
    }
    return f.h;
}

// B-fragment (32x16 f16, K x N) from row-major f16 weights W[4H][ld]:
// B[k][n] = W[col+n][k]. Lane n = N column, halves j = K = 16*hi + j ->
// 16 contiguous halves (32B, aligned) per lane -> global_load_b128 pairs.
__device__ __forceinline__ v16h load_b_frag(const _Float16* w, int ld, int col, int kbase, int lane) {
    const int n  = lane & 15;
    const int hi = lane >> 4;
    return *(const v16h*)(w + (size_t)(col + n) * ld + kbase + 16 * hi);
}

// Async-copy one BT x KIN f16 tile (row stride `rowstride` halves) from
// global into LDS buffer xh_buf (rows padded to XH_LD halves). 16B chunks,
// one async b128 per chunk; completion via ASYNCcnt.
template <int KIN>
__device__ __forceinline__ void stage_x_async(const _Float16* src, size_t rowstride,
                                              _Float16* xh_buf, int tid) {
    constexpr int XH_LD  = KIN + 8;
    constexpr int CHUNKS = BT * KIN / 8;        // 16-byte chunks
    for (int c = tid; c < CHUNKS; c += WTH) {
        const int m = c / (KIN / 8);
        const int j = c - m * (KIN / 8);
        const unsigned long long g =
            (unsigned long long)(uintptr_t)(src + (size_t)m * rowstride + (size_t)j * 8);
        // LDS aperture keeps the byte offset in the low 32 bits of the flat ptr.
        const uint32_t lds = (uint32_t)(uintptr_t)(xh_buf + m * XH_LD + j * 8);
        asm volatile("global_load_async_to_lds_b128 %0, %1, off"
                     :: "v"(lds), "v"(g) : "memory");
    }
}

__device__ __forceinline__ void wait_async_all() {
    asm volatile("s_wait_asynccnt 0x0" ::: "memory");
}

// ---------------------------------------------------------------------------
// One LSTM layer. KIN = input width (128 layer0, 512 layer1); x is f16.
// STORE_HS: layer0 writes every h_t (f16); layer1 writes only h_T (f32).
// Each wave owns 64 hidden columns (4 tiles of 16) and computes all 4 gates
// for them, so activations fuse in registers on the C-matrix layout.
// ---------------------------------------------------------------------------
template <int KIN, bool STORE_HS>
__global__ __launch_bounds__(WTH)
void lstm_layer_kernel(const _Float16* __restrict__ x,     // f16 inputs
                       size_t x_row_stride,                // halves between batch rows
                       size_t x_t_stride,                  // halves between timesteps
                       const _Float16* __restrict__ Wih,   // [4H, KIN] f16
                       const _Float16* __restrict__ Whh,   // [4H, HID] f16
                       const float*    __restrict__ bias,  // [4H]
                       _Float16*       __restrict__ hs_out,   // [T,B,HID] f16
                       float*          __restrict__ hfin_out) // [B,HID] f32
{
    constexpr int XH_LD = KIN + 8;   // pad rows: stagger LDS banks for A loads
    constexpr int HH_LD = HID + 8;
    constexpr int C_LD  = HID + 4;

    extern __shared__ char smem[];
    _Float16* xh  = (_Float16*)smem;                                            // [2][BT][XH_LD]
    _Float16* hh  = (_Float16*)(smem + sizeof(_Float16) * 2 * BT * XH_LD);      // [BT][HH_LD]
    float*    cst = (float*)(smem + sizeof(_Float16) * (2 * BT * XH_LD + BT * HH_LD)); // [BT][C_LD]

    const int tid   = threadIdx.x;
    const int wave  = tid >> 5;
    const int lane  = tid & 31;
    const int n     = lane & 15;
    const int hi    = lane >> 4;
    const int brow0 = blockIdx.x * BT;

    const _Float16* xblk = x + (size_t)brow0 * x_row_stride;  // this WG's rows

    for (int i = tid; i < BT * HH_LD; i += WTH) hh[i]  = (_Float16)0.0f;
    for (int i = tid; i < BT * C_LD;  i += WTH) cst[i] = 0.0f;

    // prologue: async-stage x_0 into buffer 0
    stage_x_async<KIN>(xblk, x_row_stride, xh, tid);
    wait_async_all();
    __syncthreads();

    for (int t = 0; t < T_STEPS; ++t) {
        const _Float16* xcur = xh + (size_t)(t & 1) * BT * XH_LD;

        // prefetch next step's x tile into the other buffer (hidden by WMMAs)
        if (t + 1 < T_STEPS) {
            stage_x_async<KIN>(xblk + (size_t)(t + 1) * x_t_stride, x_row_stride,
                               xh + (size_t)((t + 1) & 1) * BT * XH_LD, tid);
        }

        _Float16 hnew[4][8];

#pragma unroll
        for (int ht = 0; ht < 4; ++ht) {
            const int hcol0 = wave * 64 + ht * 16;   // hidden-column tile base

            v8f acc[4];
#pragma unroll
            for (int g = 0; g < 4; ++g) {
                const float bv = bias[g * HID + hcol0 + n];  // bias depends on N only
#pragma unroll
                for (int r = 0; r < 8; ++r) acc[g][r] = bv;
            }

            // gates += x_t @ Wih^T
#pragma unroll 2
            for (int k0 = 0; k0 < KIN; k0 += 32) {
                const v16h a = load_a_frag(xcur, XH_LD, k0, lane);
#pragma unroll
                for (int g = 0; g < 4; ++g) {
                    const v16h b = load_b_frag(Wih, KIN, g * HID + hcol0, k0, lane);
                    acc[g] = __builtin_amdgcn_wmma_f32_16x16x32_f16(
                        false, a, false, b, (short)0, acc[g], false, false);
                }
            }
            // gates += h_{t-1} @ Whh^T
#pragma unroll 2
            for (int k0 = 0; k0 < HID; k0 += 32) {
                const v16h a = load_a_frag(hh, HH_LD, k0, lane);
#pragma unroll
                for (int g = 0; g < 4; ++g) {
                    const v16h b = load_b_frag(Whh, HID, g * HID + hcol0, k0, lane);
                    acc[g] = __builtin_amdgcn_wmma_f32_16x16x32_f16(
                        false, a, false, b, (short)0, acc[g], false, false);
                }
            }

            // ---- fused activations on C layout: lane -> N=n, VGPR r -> M ----
#pragma unroll
            for (int r = 0; r < 8; ++r) {
                const int m   = r + 8 * hi;
                const int col = hcol0 + n;
                const float ig = sigf(acc[0][r]);
                const float fg = sigf(acc[1][r]);
                const float gg = tanhf(acc[2][r]);
                const float og = sigf(acc[3][r]);
                const float cn = fg * cst[m * C_LD + col] + ig * gg;
                const float hv = og * tanhf(cn);
                cst[m * C_LD + col] = cn;     // (m,col) owned by exactly one lane
                hnew[ht][r] = (_Float16)hv;
                if (STORE_HS) {
                    hs_out[((size_t)t * BATCH + brow0 + m) * HID + col] = (_Float16)hv;
                } else if (t == T_STEPS - 1) {
                    hfin_out[(size_t)(brow0 + m) * HID + col] = hv;
                }
            }
        }

        __syncthreads();   // all waves finished reading hh/xh for this step
#pragma unroll
        for (int ht = 0; ht < 4; ++ht) {
            const int hcol0 = wave * 64 + ht * 16;
#pragma unroll
            for (int r = 0; r < 8; ++r)
                hh[(r + 8 * hi) * HH_LD + hcol0 + n] = hnew[ht][r];
        }
        wait_async_all();  // next x tile landed in LDS (per wave)
        __syncthreads();   // new h + prefetched x visible to all waves
    }
}

// f32 -> f16 conversion, 4 elements per thread (n divisible by 4).
__global__ void cvt_f32_f16_x4_kernel(const float* __restrict__ in,
                                      _Float16* __restrict__ out, int n4) {
    const int i = blockIdx.x * blockDim.x + threadIdx.x;
    if (i < n4) {
        const float4 v = ((const float4*)in)[i];
        v4h h;
        h[0] = (_Float16)v.x; h[1] = (_Float16)v.y;
        h[2] = (_Float16)v.z; h[3] = (_Float16)v.w;
        *(v4h*)(out + (size_t)i * 4) = h;
    }
}

// out[b,o] = h_T[b,:] . W_out[o,:] + b_out[o]   (2048x512x10, trivial)
__global__ void out_gemm_kernel(const float* __restrict__ h,
                                const float* __restrict__ Wout,
                                const float* __restrict__ bout,
                                float* __restrict__ out) {
    const int idx = blockIdx.x * blockDim.x + threadIdx.x;
    if (idx >= BATCH * 10) return;
    const int b = idx / 10, o = idx - b * 10;
    const float4* hv = (const float4*)(h + (size_t)b * HID);
    const float4* wv = (const float4*)(Wout + (size_t)o * HID);
    float s = bout[o];
#pragma unroll 4
    for (int k = 0; k < HID / 4; ++k) {
        const float4 a = hv[k], w = wv[k];
        s += a.x * w.x + a.y * w.y + a.z * w.z + a.w * w.w;
    }
    out[idx] = s;
}

extern "C" void kernel_launch(void* const* d_in, const int* in_sizes, int n_in,
                              void* d_out, int out_size, void* d_ws, size_t ws_size,
                              hipStream_t stream) {
    (void)in_sizes; (void)n_in; (void)out_size; (void)ws_size;

    const float* xb    = (const float*)d_in[0]; // [B, L]
    const float* W_ih0 = (const float*)d_in[1]; // [4H, 128]
    const float* W_hh0 = (const float*)d_in[2]; // [4H, 512]
    const float* b0    = (const float*)d_in[3];
    const float* W_ih1 = (const float*)d_in[4]; // [4H, 512]
    const float* W_hh1 = (const float*)d_in[5]; // [4H, 512]
    const float* b1    = (const float*)d_in[6];
    const float* W_out = (const float*)d_in[7]; // [10, 512]
    const float* b_out = (const float*)d_in[8];
    float* out = (float*)d_out;                 // [B, 10]

    // ---- carve workspace (256B aligned) ----
    char* wsb = (char*)d_ws;
    size_t off = 0;
    auto carve = [&](size_t bytes) -> void* {
        void* p = wsb + off;
        off = (off + bytes + 255) & ~(size_t)255;
        return p;
    };
    _Float16* Wih0h = (_Float16*)carve((size_t)GATES * 128 * sizeof(_Float16));
    _Float16* Whh0h = (_Float16*)carve((size_t)GATES * HID * sizeof(_Float16));
    _Float16* Wih1h = (_Float16*)carve((size_t)GATES * HID * sizeof(_Float16));
    _Float16* Whh1h = (_Float16*)carve((size_t)GATES * HID * sizeof(_Float16));
    _Float16* xbh   = (_Float16*)carve((size_t)BATCH * T_STEPS * 128 * sizeof(_Float16)); // 32MB
    _Float16* hs0   = (_Float16*)carve((size_t)T_STEPS * BATCH * HID * sizeof(_Float16)); // 128MB
    float*    hfin  = (float*)carve((size_t)BATCH * HID * sizeof(float));

    // ---- f32 -> f16 conversions (weights + whole input) ----
    {
        const int n0 = GATES * 128 / 4, n1 = GATES * HID / 4;
        const int nx = BATCH * T_STEPS * 128 / 4;
        cvt_f32_f16_x4_kernel<<<(n0 + 255) / 256, 256, 0, stream>>>(W_ih0, Wih0h, n0);
        cvt_f32_f16_x4_kernel<<<(n1 + 255) / 256, 256, 0, stream>>>(W_hh0, Whh0h, n1);
        cvt_f32_f16_x4_kernel<<<(n1 + 255) / 256, 256, 0, stream>>>(W_ih1, Wih1h, n1);
        cvt_f32_f16_x4_kernel<<<(n1 + 255) / 256, 256, 0, stream>>>(W_hh1, Whh1h, n1);
        cvt_f32_f16_x4_kernel<<<(nx + 255) / 256, 256, 0, stream>>>(xb, xbh, nx);
    }

    const int nblk = BATCH / BT; // 128 workgroups

    // layer 0: x = xbh [B, T*128], store all h_t (f16)
    {
        const size_t smem = sizeof(_Float16) * (size_t)2 * BT * (128 + 8)
                          + sizeof(_Float16) * (size_t)BT * (HID + 8)
                          + sizeof(float)    * (size_t)BT * (HID + 4);
        lstm_layer_kernel<128, true><<<nblk, WTH, smem, stream>>>(
            xbh, (size_t)T_STEPS * 128, (size_t)128,
            Wih0h, Whh0h, b0, hs0, nullptr);
    }
    // layer 1: x = hs0 [T, B, 512], store only h_T (f32)
    {
        const size_t smem = sizeof(_Float16) * (size_t)2 * BT * (HID + 8)
                          + sizeof(_Float16) * (size_t)BT * (HID + 8)
                          + sizeof(float)    * (size_t)BT * (HID + 4);
        lstm_layer_kernel<512, false><<<nblk, WTH, smem, stream>>>(
            hs0, (size_t)HID, (size_t)BATCH * HID,
            Wih1h, Whh1h, b1, nullptr, hfin);
    }

    out_gemm_kernel<<<(BATCH * 10 + 255) / 256, 256, 0, stream>>>(hfin, W_out, b_out, out);
}
